// PyramidVisionMamba_65661460021384
// MI455X (gfx1250) — matrix operand order repaired and checked by reading, not compiled, not run on hardware
//
#include <hip/hip_runtime.h>
#include <cmath>

// ---------------------------------------------------------------------------
// PyramidVisionMamba forward for MI455X (gfx1250, wave32, WMMA).
// Dense projections use a generic f16-WMMA GEMM (f32 accumulate) with
// hoisted row pointers, batched b128 tile loads and f32->f16 conversion
// into LDS. Activations stay fp32 in HBM (bandwidth-bound at 23.3 TB/s).
// ---------------------------------------------------------------------------

typedef __attribute__((ext_vector_type(16))) _Float16 v16h;
typedef __attribute__((ext_vector_type(4)))  _Float16 v4h;
typedef __attribute__((ext_vector_type(8)))  float    v8f;

#define TILE_M 128
#define TILE_N 128
#define TILE_K 32

__device__ __forceinline__ float silu_f(float x)     { return x / (1.f + __expf(-x)); }
__device__ __forceinline__ float softplus_f(float x) { return (x > 20.f) ? x : log1pf(__expf(x)); }

// C[m, coff+n] = sum_k A[m*lda + aoff + k] * W[n*ldw + k]
//             (+ bias[n]) (+ (*res_scale) * RES[m*ldr + roff + n]) (act: 1=softplus)
// Requirements (satisfied by every call site): lda, ldw, K multiples of 4;
// A/W row starts 16B aligned (aoff multiple of 4).
__global__ __launch_bounds__(256)
void gemm_wmma_kernel(const float* __restrict__ A, int lda, int aoff,
                      const float* __restrict__ W, int ldw,
                      const float* __restrict__ bias,
                      const float* __restrict__ RES, int ldr, int roff,
                      const float* __restrict__ res_scale,
                      float* __restrict__ C, int ldc, int coff,
                      int M, int N, int K, int act)
{
    __shared__ _Float16 sA[TILE_M * TILE_K];
    __shared__ _Float16 sB[TILE_N * TILE_K];

    const int t    = threadIdx.x;
    const int wave = t >> 5;
    const int lane = t & 31;
    const int m0 = blockIdx.y * TILE_M;
    const int n0 = blockIdx.x * TILE_N;
    const int wm = (wave & 3) * 32;   // 4 M-groups of 32 rows
    const int wn = (wave >> 2) * 64;  // 2 N-groups of 64 cols

    v8f acc[2][4];
#pragma unroll
    for (int mi = 0; mi < 2; ++mi)
#pragma unroll
        for (int ni = 0; ni < 4; ++ni)
#pragma unroll
            for (int e = 0; e < 8; ++e) acc[mi][ni][e] = 0.f;

    // ---- Loop-invariant tile-load geometry: thread t always loads column
    // kt = (t&7)*4 of the tile, rows (t>>3) + 32*i, i = 0..3.
    const int kt    = (t & 7) * 4;
    const int rbase = t >> 3;
    const float* rowA[4];
    const float* rowB[4];
    bool mA[4], mB[4];
#pragma unroll
    for (int i = 0; i < 4; ++i) {
        const int gm = m0 + rbase + 32 * i;
        const int gn = n0 + rbase + 32 * i;
        mA[i] = gm < M;
        mB[i] = gn < N;
        rowA[i] = A + (long)(mA[i] ? gm : m0) * lda + aoff;   // clamped, always valid
        rowB[i] = W + (long)(mB[i] ? gn : n0) * ldw;
    }
    _Float16* const sAp = &sA[t * 4];   // == row(rbase)*TILE_K + kt, +1024 per i
    _Float16* const sBp = &sB[t * 4];

    const int lrow  = lane & 15;
    const int ahalf = (lane < 16) ? 0 : 16;   // contiguous K halves, matches B
    const int bhalf = (lane < 16) ? 0 : 16;   // contiguous K halves, matches A

    for (int k0 = 0; k0 < K; k0 += TILE_K) {
        const bool kin = (k0 + kt) < K;        // K%4==0 -> whole chunk in/out
        const int  ko  = kin ? (k0 + kt) : 0;

        // Phase 1: 8 independent global_load_b128 (batched under one wait)
        float4 va[4], vb[4];
#pragma unroll
        for (int i = 0; i < 4; ++i) {
            va[i] = *(const float4*)(rowA[i] + ko);
            vb[i] = *(const float4*)(rowB[i] + ko);
        }
        if (k0 + TILE_K < K)   // global_prefetch_b8 of next A tile
            __builtin_prefetch(rowA[0] + kt + k0 + TILE_K, 0, 1);

        // Phase 2: cndmask OOB to 0, cvt f32->f16, ds_store_b64
#pragma unroll
        for (int i = 0; i < 4; ++i) {
            const bool ia = mA[i] && kin;
            const bool ib = mB[i] && kin;
            v4h ha, hb;
            ha[0] = (_Float16)(ia ? va[i].x : 0.f);
            ha[1] = (_Float16)(ia ? va[i].y : 0.f);
            ha[2] = (_Float16)(ia ? va[i].z : 0.f);
            ha[3] = (_Float16)(ia ? va[i].w : 0.f);
            hb[0] = (_Float16)(ib ? vb[i].x : 0.f);
            hb[1] = (_Float16)(ib ? vb[i].y : 0.f);
            hb[2] = (_Float16)(ib ? vb[i].z : 0.f);
            hb[3] = (_Float16)(ib ? vb[i].w : 0.f);
            *(v4h*)(sAp + i * 1024) = ha;
            *(v4h*)(sBp + i * 1024) = hb;
        }
        __syncthreads();

        // A fragments: lane holds row (lane&15); lanes<16: K 0-15; lanes>=16: K 16-31
        v16h af[2];
#pragma unroll
        for (int mi = 0; mi < 2; ++mi) {
            const _Float16* ap = &sA[(wm + mi * 16 + lrow) * TILE_K];
#pragma unroll
            for (int j = 0; j < 16; ++j) {
                af[mi][j]     = ap[ahalf + j];
            }
        }
        // B fragments: lane holds column (lane&15); lanes<16: K 0-15; lanes>=16: K 16-31
        v16h bf[4];
#pragma unroll
        for (int ni = 0; ni < 4; ++ni) {
            const _Float16* bp = &sB[(wn + ni * 16 + lrow) * TILE_K + bhalf];
#pragma unroll
            for (int j = 0; j < 16; ++j) bf[ni][j] = bp[j];
        }
#pragma unroll
        for (int mi = 0; mi < 2; ++mi)
#pragma unroll
            for (int ni = 0; ni < 4; ++ni)
                acc[mi][ni] = __builtin_amdgcn_wmma_f32_16x16x32_f16(
                    false, af[mi], false, bf[ni], (short)0, acc[mi][ni], false, false);
        __syncthreads();
    }

    const float rscale = res_scale ? *res_scale : 0.f;
    const bool interior = (m0 + TILE_M <= M) && (n0 + TILE_N <= N);
    // C/D 16x16 f32 layout: elem e -> row e (+8 for lanes>=16), col = lane&15
    if (interior) {
#pragma unroll
        for (int mi = 0; mi < 2; ++mi)
#pragma unroll
            for (int ni = 0; ni < 4; ++ni)
#pragma unroll
                for (int e = 0; e < 8; ++e) {
                    int row = m0 + wm + mi * 16 + e + ((lane >= 16) ? 8 : 0);
                    int col = n0 + wn + ni * 16 + lrow;
                    float v = acc[mi][ni][e];
                    if (bias) v += bias[col];
                    if (RES)  v += rscale * RES[(long)row * ldr + roff + col];
                    if (act == 1) v = softplus_f(v);
                    C[(long)row * ldc + coff + col] = v;
                }
    } else {
#pragma unroll
        for (int mi = 0; mi < 2; ++mi)
#pragma unroll
            for (int ni = 0; ni < 4; ++ni)
#pragma unroll
                for (int e = 0; e < 8; ++e) {
                    int row = m0 + wm + mi * 16 + e + ((lane >= 16) ? 8 : 0);
                    int col = n0 + wn + ni * 16 + lrow;
                    if (row < M && col < N) {
                        float v = acc[mi][ni][e];
                        if (bias) v += bias[col];
                        if (RES)  v += rscale * RES[(long)row * ldr + roff + col];
                        if (act == 1) v = softplus_f(v);
                        C[(long)row * ldc + coff + col] = v;
                    }
                }
    }
}

// Patch embed: 4x4/4 conv 3->512, NCHW input -> token-major (B*Hp*Wp, 512) + bias
__global__ __launch_bounds__(256)
void patch_embed_kernel(const float* __restrict__ x, const float* __restrict__ pw,
                        const float* __restrict__ pb, float* __restrict__ out,
                        int Hp, int Wp, int IMG)
{
    __shared__ float px[48];   // 3*4*4 input patch shared across 512 outputs
    int pidx = blockIdx.x;
    int b = pidx / (Hp * Wp);
    int hw = pidx % (Hp * Wp);
    int h = hw / Wp, w = hw % Wp;
    int tid = threadIdx.x;
    if (tid < 48) {
        int c = tid >> 4, kh = (tid >> 2) & 3, kw = tid & 3;
        px[tid] = x[((b * 3 + c) * IMG + (h * 4 + kh)) * IMG + (w * 4 + kw)];
    }
    __syncthreads();
    for (int o = tid; o < 512; o += 256) {
        float acc = pb[o];
        const float* wp = pw + o * 48;
#pragma unroll
        for (int j = 0; j < 48; ++j) acc = fmaf(wp[j], px[j], acc);
        out[pidx * 512 + o] = acc;
    }
}

// LayerNorm over last dim Cn (512 or 2048); one block per row
__global__ __launch_bounds__(256)
void layernorm_kernel(const float* __restrict__ x, const float* __restrict__ w,
                      const float* __restrict__ b, float* __restrict__ y, int Cn)
{
    __shared__ float sh[512];
    int row = blockIdx.x;
    const float* xr = x + (long)row * Cn;
    float s = 0.f, s2 = 0.f;
    for (int c = threadIdx.x; c < Cn; c += 256) { float v = xr[c]; s += v; s2 += v * v; }
    sh[threadIdx.x] = s; sh[256 + threadIdx.x] = s2;
    __syncthreads();
    for (int st = 128; st > 0; st >>= 1) {
        if (threadIdx.x < st) {
            sh[threadIdx.x]       += sh[threadIdx.x + st];
            sh[256 + threadIdx.x] += sh[256 + threadIdx.x + st];
        }
        __syncthreads();
    }
    float mean = sh[0] / Cn;
    float var  = sh[256] / Cn - mean * mean;
    float rs = rsqrtf(var + 1e-5f);
    float* yr = y + (long)row * Cn;
    for (int c = threadIdx.x; c < Cn; c += 256) yr[c] = (xr[c] - mean) * rs * w[c] + b[c];
}

// Causal depthwise conv (k=4) over L within each batch, on x half of xz, + SiLU
__global__ __launch_bounds__(256)
void dwconv_silu_kernel(const float* __restrict__ xz, const float* __restrict__ cw,
                        const float* __restrict__ cb, float* __restrict__ out, int L)
{
    int idx = blockIdx.x * 256 + threadIdx.x;
    int d = idx & 255;
    int tok = idx >> 8;
    int l = tok % L;
    float acc = cb[d];
#pragma unroll
    for (int k = 0; k < 4; ++k) {
        int li = l + k - 3;
        if (li >= 0) acc = fmaf(cw[d * 4 + k], xz[(tok + k - 3) * 512 + d], acc);
    }
    out[tok * 256 + d] = silu_f(acc);
}

// Selective scan: block = batch, thread = channel d; 16 states in registers
__global__ __launch_bounds__(256)
void ssm_scan_kernel(const float* __restrict__ dt, const float* __restrict__ xc,
                     const float* __restrict__ dbc, const float* __restrict__ A_log,
                     float* __restrict__ ys, int L)
{
    int b = blockIdx.x;
    int d = threadIdx.x;
    float A[16], h[16];
#pragma unroll
    for (int n = 0; n < 16; ++n) { A[n] = -__expf(A_log[d * 16 + n]); h[n] = 0.f; }
    int base = b * L;
    for (int l = 0; l < L; ++l) {
        int tok = base + l;
        float dtv = dt[tok * 256 + d];
        float xv  = xc[tok * 256 + d];
        const float* bc = dbc + tok * 40;   // [0:8)=dt_raw, [8:24)=B, [24:40)=C
        float dx = dtv * xv;
        float y = 0.f;
#pragma unroll
        for (int n = 0; n < 16; ++n) {
            h[n] = h[n] * __expf(dtv * A[n]) + dx * bc[8 + n];
            y = fmaf(h[n], bc[24 + n], y);
        }
        ys[tok * 256 + d] = y;
    }
}

// y = (ys + xc*D) * silu(z)  (z = cols [256,512) of xz), in place on ys
__global__ __launch_bounds__(256)
void gate_kernel(float* __restrict__ ys, const float* __restrict__ xc,
                 const float* __restrict__ xz, const float* __restrict__ D)
{
    int idx = blockIdx.x * 256 + threadIdx.x;
    int d = idx & 255, tok = idx >> 8;
    float y = ys[tok * 256 + d] + xc[tok * 256 + d] * D[d];
    float z = xz[tok * 512 + 256 + d];
    ys[tok * 256 + d] = y * silu_f(z);
}

// 2x2 space-to-depth: (B,Hs,Hs,512) -> (B*H2*H2, 2048), concat order x0,x1,x2,x3
__global__ __launch_bounds__(256)
void merge_gather_kernel(const float* __restrict__ x, float* __restrict__ xc,
                         int Hs, int H2)
{
    long idx = (long)blockIdx.x * 256 + threadIdx.x;
    long total = (long)8 * H2 * H2 * 2048;
    if (idx >= total) return;
    int c = (int)(idx & 2047);
    long pos = idx >> 11;
    int w2 = (int)(pos % H2);
    int h2 = (int)((pos / H2) % H2);
    int b  = (int)(pos / ((long)H2 * H2));
    int quad = c >> 9, ci = c & 511;
    int dh = quad & 1, dw = quad >> 1;
    xc[idx] = x[(((long)(b * Hs + h2 * 2 + dh)) * Hs + (w2 * 2 + dw)) * 512 + ci];
}

// Bilinear upsample 11x11 -> 352x352 per (b), matching the reference matrices
__global__ __launch_bounds__(256)
void bilinear_kernel(const float* __restrict__ o, float* __restrict__ out)
{
    int idx = blockIdx.x * 256 + threadIdx.x;
    const int NPIX = 352 * 352;
    if (idx >= 8 * NPIX) return;
    int q = idx % 352;
    int p = (idx / 352) % 352;
    int b = idx / NPIX;
    float ph = p * (10.f / 351.f);
    float pw = q * (10.f / 351.f);
    int lh = (int)floorf(ph); lh = lh < 0 ? 0 : (lh > 9 ? 9 : lh);
    int lw = (int)floorf(pw); lw = lw < 0 ? 0 : (lw > 9 ? 9 : lw);
    float fh = ph - lh, fw = pw - lw;
    const float* ob = o + b * 121;
    float v00 = ob[lh * 11 + lw],       v10 = ob[(lh + 1) * 11 + lw];
    float v01 = ob[lh * 11 + lw + 1],   v11 = ob[(lh + 1) * 11 + lw + 1];
    out[idx] = (1 - fh) * (1 - fw) * v00 + fh * (1 - fw) * v10
             + (1 - fh) * fw * v01 + fh * fw * v11;
}

// ---------------------------------------------------------------------------

static inline void launch_gemm(hipStream_t st,
                               const float* A, int lda, int aoff,
                               const float* W, int ldw, const float* bias,
                               const float* RES, int ldr, int roff, const float* rs,
                               float* C, int ldc, int coff,
                               int M, int N, int K, int act)
{
    dim3 g((N + TILE_N - 1) / TILE_N, (M + TILE_M - 1) / TILE_M);
    gemm_wmma_kernel<<<g, 256, 0, st>>>(A, lda, aoff, W, ldw, bias,
                                        RES, ldr, roff, rs, C, ldc, coff, M, N, K, act);
}

extern "C" void kernel_launch(void* const* d_in, const int* in_sizes, int n_in,
                              void* d_out, int out_size, void* d_ws, size_t ws_size,
                              hipStream_t stream)
{
    (void)in_sizes; (void)n_in; (void)out_size; (void)ws_size;

    // Leaf order = setup_inputs() dict insertion order
    const float* x       = (const float*)d_in[0];
    const float* patch_w = (const float*)d_in[1];
    const float* patch_b = (const float*)d_in[2];
    const float* pn_w    = (const float*)d_in[3];
    const float* pn_b    = (const float*)d_in[4];

    struct Blk {
        const float *norm_w, *norm_b, *in_proj, *conv_w, *conv_b, *x_proj,
                    *dt_w, *dt_b, *A_log, *D, *out_proj, *proj_w, *proj_b, *skip;
    } blk[4];
    int idx = 5;
    for (int i = 0; i < 4; ++i) {
        blk[i].norm_w   = (const float*)d_in[idx + 0];
        blk[i].norm_b   = (const float*)d_in[idx + 1];
        blk[i].in_proj  = (const float*)d_in[idx + 2];
        blk[i].conv_w   = (const float*)d_in[idx + 3];
        blk[i].conv_b   = (const float*)d_in[idx + 4];
        blk[i].x_proj   = (const float*)d_in[idx + 5];
        blk[i].dt_w     = (const float*)d_in[idx + 6];
        blk[i].dt_b     = (const float*)d_in[idx + 7];
        blk[i].A_log    = (const float*)d_in[idx + 8];
        blk[i].D        = (const float*)d_in[idx + 9];
        blk[i].out_proj = (const float*)d_in[idx + 10];
        blk[i].proj_w   = (const float*)d_in[idx + 11];
        blk[i].proj_b   = (const float*)d_in[idx + 12];
        blk[i].skip     = (const float*)d_in[idx + 13];
        idx += 14;
    }
    const float *mg_nw[3], *mg_nb[3], *mg_rw[3];
    for (int i = 0; i < 3; ++i) {
        mg_nw[i] = (const float*)d_in[idx++];
        mg_nb[i] = (const float*)d_in[idx++];
        mg_rw[i] = (const float*)d_in[idx++];
    }
    const float* head_w = (const float*)d_in[idx++];
    const float* out_w  = (const float*)d_in[idx++];
    const float* out_b  = (const float*)d_in[idx++];

    // Workspace layout (fp32), T0 = 8*88*88 tokens
    const size_t T0 = 8 * 88 * 88;
    float* ws      = (float*)d_ws;
    float* big0    = ws;
    float* big1    = big0    + T0 * 512;
    float* big2    = big1    + T0 * 512;
    float* bufXZ   = big2    + T0 * 512;   // also reused as merge gather / head scratch
    float* bufConv = bufXZ   + T0 * 512;
    float* bufDBC  = bufConv + T0 * 256;
    float* bufDT   = bufDBC  + T0 * 40;
    float* bufYS   = bufDT   + T0 * 256;

    // Patch embed -> big0 ; patch LN -> big1
    patch_embed_kernel<<<(int)T0, 256, 0, stream>>>(x, patch_w, patch_b, big0, 88, 88, 352);
    layernorm_kernel<<<(int)T0, 256, 0, stream>>>(big0, pn_w, pn_b, big1, 512);

    float* act = big1;
    float* s1  = big0;
    float* s2  = big2;

    for (int s = 0; s < 4; ++s) {
        int Hs = 88 >> s;
        int L  = Hs * Hs;
        int T  = 8 * L;
        const Blk& bp = blk[s];

        // xn = LN(act) in s1
        layernorm_kernel<<<T, 256, 0, stream>>>(act, bp.norm_w, bp.norm_b, s1, 512);

        // 4 Mamba chunks (columns c*128..c*128+127 of s1), outputs -> s2 (xm)
        for (int c = 0; c < 4; ++c) {
            launch_gemm(stream, s1, 512, c * 128, bp.in_proj, 128, nullptr,
                        nullptr, 0, 0, nullptr, bufXZ, 512, 0, T, 512, 128, 0);
            dwconv_silu_kernel<<<T, 256, 0, stream>>>(bufXZ, bp.conv_w, bp.conv_b, bufConv, L);
            launch_gemm(stream, bufConv, 256, 0, bp.x_proj, 256, nullptr,
                        nullptr, 0, 0, nullptr, bufDBC, 40, 0, T, 40, 256, 0);
            launch_gemm(stream, bufDBC, 40, 0, bp.dt_w, 8, bp.dt_b,
                        nullptr, 0, 0, nullptr, bufDT, 256, 0, T, 256, 8, /*softplus*/1);
            ssm_scan_kernel<<<8, 256, 0, stream>>>(bufDT, bufConv, bufDBC, bp.A_log, bufYS, L);
            gate_kernel<<<T, 256, 0, stream>>>(bufYS, bufConv, bufXZ, bp.D);
            launch_gemm(stream, bufYS, 256, 0, bp.out_proj, 256, nullptr,
                        s1, 512, c * 128, bp.skip,       // + skip_scale * chunk
                        s2, 512, c * 128, T, 128, 256, 0);
        }

        // LN(xm) -> act (block input dead), proj GEMM -> s1, rotate buffers
        layernorm_kernel<<<T, 256, 0, stream>>>(s2, bp.norm_w, bp.norm_b, act, 512);
        launch_gemm(stream, act, 512, 0, bp.proj_w, 512, bp.proj_b,
                    nullptr, 0, 0, nullptr, s1, 512, 0, T, 512, 512, 0);
        { float* na = s1; s1 = s2; s2 = act; act = na; }

        if (s < 3) {
            int H2 = Hs / 2;
            int Tq = 8 * H2 * H2;
            long tot = (long)Tq * 2048;
            merge_gather_kernel<<<(int)((tot + 255) / 256), 256, 0, stream>>>(act, bufXZ, Hs, H2);
            layernorm_kernel<<<Tq, 256, 0, stream>>>(bufXZ, mg_nw[s], mg_nb[s], s1, 2048);
            launch_gemm(stream, s1, 2048, 0, mg_rw[s], 2048, nullptr,
                        nullptr, 0, 0, nullptr, s2, 512, 0, Tq, 512, 2048, 0);
            { float* na = s2; s2 = act; act = na; }
        }
    }

    // Head: (968,512) -> (968,32) -> (968,1), then bilinear 11->352
    const int T3 = 8 * 11 * 11;
    float* hb = bufXZ;
    float* ob = bufConv;
    launch_gemm(stream, act, 512, 0, head_w, 512, nullptr,
                nullptr, 0, 0, nullptr, hb, 32, 0, T3, 32, 512, 0);
    launch_gemm(stream, hb, 32, 0, out_w, 32, out_b,
                nullptr, 0, 0, nullptr, ob, 1, 0, T3, 1, 32, 0);
    bilinear_kernel<<<(8 * 352 * 352 + 255) / 256, 256, 0, stream>>>(ob, (float*)d_out);
}
